// Model_46660524703806
// MI455X (gfx1250) — matrix operand order repaired and checked by reading, not compiled, not run on hardware
//
#include <hip/hip_runtime.h>
#include <hip/hip_bf16.h>

// ---- problem constants (match reference) ----
static constexpr int Bb = 16, Ll = 512, Cc = 32;
static constexpr int Pp = 64, Dd = 128, DFFc = 256;
static constexpr int Hh = 8, DHDc = 16;
static constexpr int PREDc = 128, KCc = 2, Kctx = 2 * KCc + 1;
static constexpr int Dhc = Dd / 2;            // 64
static constexpr int NLc = 2;
static constexpr int HPc = 128, HCc = 64;

typedef __attribute__((ext_vector_type(16))) __bf16 v16bf;
typedef __attribute__((ext_vector_type(8)))  float  v8f;

__device__ __forceinline__ float gelu_f(float x) {
    return 0.5f * x * (1.0f + erff(x * 0.70710678118654752f));
}
__device__ __forceinline__ float wred_sum(float s) {
    #pragma unroll
    for (int o = 16; o; o >>= 1) s += __shfl_xor(s, o, 32);
    return s;
}

// ---------------- scalars: min_t, pw ----------------
__global__ void scalars_kernel(const float* __restrict__ x_mark,
                               const float* __restrict__ y_mark, float* scal) {
    __shared__ float smn[256], smx[256];
    float mn = 1e30f, mx = -1e30f;
    for (int i = threadIdx.x; i < Bb * Ll; i += 256) mn = fminf(mn, x_mark[i]);
    for (int i = threadIdx.x; i < Bb * PREDc; i += 256) mx = fmaxf(mx, y_mark[i]);
    smn[threadIdx.x] = mn; smx[threadIdx.x] = mx; __syncthreads();
    for (int s = 128; s; s >>= 1) {
        if (threadIdx.x < s) {
            smn[threadIdx.x] = fminf(smn[threadIdx.x], smn[threadIdx.x + s]);
            smx[threadIdx.x] = fmaxf(smx[threadIdx.x], smx[threadIdx.x + s]);
        }
        __syncthreads();
    }
    if (threadIdx.x == 0) {
        float mint = smn[0], maxt = smx[0];
        if (maxt == mint) maxt = mint + 1.0f;
        scal[0] = mint;
        scal[1] = (maxt - mint + 1e-8f) / (float)Pp;
    }
}

// ---------------- per-(b,p) segment boundaries (x_mark sorted along L) ----------------
__global__ void bounds_kernel(const float* __restrict__ x_mark,
                              const float* __restrict__ scal, int* __restrict__ startp) {
    int idx = blockIdx.x * 256 + threadIdx.x;
    if (idx >= Bb * (Pp + 1)) return;
    int b = idx / (Pp + 1), p = idx % (Pp + 1);
    int r;
    if (p == 0) r = 0;
    else if (p == Pp) r = Ll;
    else {
        float thr = scal[0] + scal[1] * (float)p;
        int lo = 0, hi = Ll;
        while (lo < hi) { int mid = (lo + hi) >> 1; if (x_mark[b * Ll + mid] < thr) lo = mid + 1; else hi = mid; }
        r = lo;
    }
    startp[idx] = r;
}

// ---------------- u,v: collapsed value-embedding path ----------------
__global__ void uv_kernel(const float* __restrict__ fus_w, const float* __restrict__ fus_b,
                          const float* __restrict__ val_w, const float* __restrict__ val_b,
                          float* __restrict__ u, float* __restrict__ v) {
    int d = threadIdx.x;  // 128
    float uu = 0.f, vv = fus_b[d];
    for (int j = 0; j < Dhc; ++j) { uu += fus_w[d * Dd + j] * val_w[j]; vv += fus_w[d * Dd + j] * val_b[j]; }
    u[d] = uu; v[d] = vv;
}

// ---------------- te_lin[b,l,:] = fus_w[:,64:128] @ tmMLP(sinenc(t)) ----------------
__global__ __launch_bounds__(128) void te_kernel(
    const float* __restrict__ x_mark, const float* __restrict__ inv_freq,
    const float* __restrict__ tm_w1, const float* __restrict__ tm_b1,
    const float* __restrict__ tm_w2, const float* __restrict__ tm_b2,
    const float* __restrict__ fus_w, float* __restrict__ te_lin) {
    int row = blockIdx.x;                      // b*L+l
    float t = x_mark[row];
    __shared__ float enc[64], hid[64], o64[64];
    int tid = threadIdx.x;
    if (tid < 32) { float e = t * inv_freq[tid]; enc[tid] = sinf(e); enc[tid + 32] = cosf(e); }
    __syncthreads();
    if (tid < 64) { float a = tm_b1[tid]; for (int j = 0; j < 64; ++j) a += tm_w1[tid * 64 + j] * enc[j]; hid[tid] = gelu_f(a); }
    __syncthreads();
    if (tid < 64) { float a = tm_b2[tid]; for (int j = 0; j < 64; ++j) a += tm_w2[tid * 64 + j] * hid[j]; o64[tid] = a; }
    __syncthreads();
    float a = 0.f;
    for (int j = 0; j < 64; ++j) a += fus_w[tid * Dd + 64 + j] * o64[j];
    te_lin[(size_t)row * Dd + tid] = a;
}

// ---------------- Aq[p,h,:] = wk_h^T qh[p,h], c0[p,h] = qh.bk_h ----------------
__global__ __launch_bounds__(128) void qk_kernel(
    const float* __restrict__ patch_q, const float* __restrict__ wq, const float* __restrict__ bq,
    const float* __restrict__ wk, const float* __restrict__ bk,
    float* __restrict__ Aq, float* __restrict__ c0) {
    int p = blockIdx.x >> 3, h = blockIdx.x & 7;
    __shared__ float qh[16];
    int tid = threadIdx.x;
    const float scale = 0.25f;  // 1/sqrt(DHD=16)
    if (tid < 16) {
        float a = bq[h * 16 + tid];
        for (int e = 0; e < Dd; ++e) a += wq[(h * 16 + tid) * Dd + e] * patch_q[p * Dd + e];
        qh[tid] = a * scale;
    }
    __syncthreads();
    float a = 0.f;
    for (int d = 0; d < 16; ++d) a += qh[d] * wk[(h * 16 + d) * Dd + tid];
    Aq[(p * Hh + h) * Dd + tid] = a;
    if (tid == 0) { float cc = 0.f; for (int d = 0; d < 16; ++d) cc += qh[d] * bk[h * 16 + d]; c0[p * Hh + h] = cc; }
}

// ---------------- attention pooling: one wave per (b,c,p) group ----------------
__device__ __forceinline__ void compute_pe4(
    float xval, const float* __restrict__ te_row,
    const float* u4, const float* v4, const float* fg, const float* fb,
    int d0, float* pe) {
    float ge[4], s = 0.f;
    #pragma unroll
    for (int i = 0; i < 4; ++i) {
        float pre = xval * u4[i] + v4[i] + te_row[d0 + i];
        ge[i] = gelu_f(pre); s += ge[i];
    }
    s = wred_sum(s);
    float m = s * (1.f / 128.f), q = 0.f;
    #pragma unroll
    for (int i = 0; i < 4; ++i) { float dx = ge[i] - m; q += dx * dx; }
    q = wred_sum(q);
    float inv = rsqrtf(q * (1.f / 128.f) + 1e-5f);
    #pragma unroll
    for (int i = 0; i < 4; ++i) pe[i] = (ge[i] - m) * inv * fg[i] + fb[i];
}

__global__ __launch_bounds__(256) void pool_kernel(
    const float* __restrict__ x, const unsigned char* __restrict__ x_mask,
    const float* __restrict__ te_lin,
    const float* __restrict__ u, const float* __restrict__ v,
    const float* __restrict__ fus_g, const float* __restrict__ fus_be,
    const float* __restrict__ Aq, const float* __restrict__ c0,
    const float* __restrict__ wv, const float* __restrict__ bv,
    const float* __restrict__ wo, const float* __restrict__ bo,
    const float* __restrict__ empty, const int* __restrict__ startp,
    float* __restrict__ xm) {
    __shared__ float sAll[8][Hh][Dd];   // 32 KB
    __shared__ float vcat[8][Dd];       // 4 KB
    int wid = threadIdx.x >> 5, lane = threadIdx.x & 31;
    int g = blockIdx.x * 8 + wid;
    int b = g / (Cc * Pp), c = (g / Pp) % Cc, p = g % Pp;
    int d0 = lane * 4;
    float u4[4], v4[4], fg[4], fb[4], Areg[Hh][4], c0r[Hh];
    #pragma unroll
    for (int i = 0; i < 4; ++i) { u4[i] = u[d0 + i]; v4[i] = v[d0 + i]; fg[i] = fus_g[d0 + i]; fb[i] = fus_be[d0 + i]; }
    #pragma unroll
    for (int h = 0; h < Hh; ++h) {
        c0r[h] = c0[p * Hh + h];
        #pragma unroll
        for (int i = 0; i < 4; ++i) Areg[h][i] = Aq[(p * Hh + h) * Dd + d0 + i];
    }
    int ls = startp[b * (Pp + 1) + p], le = startp[b * (Pp + 1) + p + 1];

    // pass 1: per-head max over valid members
    float mx[Hh];
    #pragma unroll
    for (int h = 0; h < Hh; ++h) mx[h] = -1e30f;
    int cnt = 0;
    for (int l = ls; l < le; ++l) {
        if (!x_mask[(size_t)(b * Ll + l) * Cc + c]) continue;
        ++cnt;
        float pe[4];
        compute_pe4(x[(size_t)(b * Ll + l) * Cc + c], te_lin + (size_t)(b * Ll + l) * Dd, u4, v4, fg, fb, d0, pe);
        #pragma unroll
        for (int h = 0; h < Hh; ++h) {
            float ps = pe[0] * Areg[h][0] + pe[1] * Areg[h][1] + pe[2] * Areg[h][2] + pe[3] * Areg[h][3];
            ps = wred_sum(ps) + c0r[h];
            mx[h] = fmaxf(mx[h], ps);
        }
    }
    // pass 2: softmax-weighted sum of pe per head
    float Wsum[Hh];
    #pragma unroll
    for (int h = 0; h < Hh; ++h) {
        Wsum[h] = 0.f;
        #pragma unroll
        for (int i = 0; i < 4; ++i) sAll[wid][h][d0 + i] = 0.f;
    }
    for (int l = ls; l < le; ++l) {
        if (!x_mask[(size_t)(b * Ll + l) * Cc + c]) continue;
        float pe[4];
        compute_pe4(x[(size_t)(b * Ll + l) * Cc + c], te_lin + (size_t)(b * Ll + l) * Dd, u4, v4, fg, fb, d0, pe);
        #pragma unroll
        for (int h = 0; h < Hh; ++h) {
            float ps = pe[0] * Areg[h][0] + pe[1] * Areg[h][1] + pe[2] * Areg[h][2] + pe[3] * Areg[h][3];
            ps = wred_sum(ps) + c0r[h];
            float w = expf(ps - mx[h]);
            Wsum[h] += w;
            #pragma unroll
            for (int i = 0; i < 4; ++i) sAll[wid][h][d0 + i] += w * pe[i];
        }
    }
    __syncthreads();
    // vcat[d] = (wv_h @ s_h)/den + bv[d]   (h = d/16)
    #pragma unroll
    for (int i = 0; i < 4; ++i) {
        int d = d0 + i, h = d >> 4;
        float den = fmaxf(Wsum[h], 1e-9f), acc = 0.f;
        for (int k = 0; k < Dd; ++k) acc += wv[d * Dd + k] * sAll[wid][h][k];
        vcat[wid][d] = acc / den + bv[d];
    }
    __syncthreads();
    #pragma unroll
    for (int i = 0; i < 4; ++i) {
        int d = d0 + i;
        float acc = bo[d];
        for (int k = 0; k < Dd; ++k) acc += wo[d * Dd + k] * vcat[wid][k];
        xm[(size_t)g * Dd + d] = (cnt > 0) ? acc : empty[d];
    }
}

// ---------------- LayerNorm over last dim 128, one wave per row ----------------
__global__ __launch_bounds__(256) void ln128_kernel(
    const float* __restrict__ in, float* __restrict__ out,
    const float* __restrict__ g, const float* __restrict__ b, int rows) {
    int wave = (blockIdx.x * 256 + threadIdx.x) >> 5, lane = threadIdx.x & 31;
    if (wave >= rows) return;
    const float4 xv = *(const float4*)(in + (size_t)wave * Dd + lane * 4);
    float s = wred_sum(xv.x + xv.y + xv.z + xv.w);
    float m = s * (1.f / 128.f);
    float dx = xv.x - m, dy = xv.y - m, dz = xv.z - m, dw = xv.w - m;
    float q = wred_sum(dx * dx + dy * dy + dz * dz + dw * dw);
    float inv = rsqrtf(q * (1.f / 128.f) + 1e-5f);
    const float4 gv = *(const float4*)(g + lane * 4);
    const float4 bv = *(const float4*)(b + lane * 4);
    float4 ov;
    ov.x = dx * inv * gv.x + bv.x; ov.y = dy * inv * gv.y + bv.y;
    ov.z = dz * inv * gv.z + bv.z; ov.w = dw * inv * gv.w + bv.w;
    *(float4*)(out + (size_t)wave * Dd + lane * 4) = ov;
}

// ---------------- WMMA bf16 GEMM: Y = act(A @ W^T + bias) (+res) ----------------
// A: MxK f32 row-major, W: NxK f32 row-major.
// One wave computes a 16 x (16*NT) strip: the A fragment is loaded once per
// K-step and reused across NT back-to-back WMMAs (4x less A traffic, and the
// independent WMMA issues let the XDL pipe overlap with the bf16 cvt VALU work).
// A fragment per ISA 16-bit 16x32 table; B fragment uses the same K table
// order (lane = column).
template <int K, int NT>
__global__ __launch_bounds__(256) void gemm_bf16_wmma(
    const float* __restrict__ A, const float* __restrict__ W,
    const float* __restrict__ bias, const float* __restrict__ res,
    float* __restrict__ Y, int M, int N, int act) {
    const int wave = (blockIdx.x * 256 + threadIdx.x) >> 5;
    const int lane = threadIdx.x & 31;
    const int groupsN = N / (16 * NT);
    const int tm = (wave / groupsN) << 4;
    const int tn = (wave % groupsN) * (16 * NT);
    if (tm >= M) return;
    const int r16 = lane & 15;
    const int hi = lane >> 4;
    const int hi8 = hi << 3;
    const float* arow = A + (size_t)(tm + r16) * K;
    v8f acc[NT];
    #pragma unroll
    for (int nt = 0; nt < NT; ++nt) {
        const float bval = bias[tn + nt * 16 + r16];
        #pragma unroll
        for (int r = 0; r < 8; ++r) acc[nt][r] = bval;
    }
    #pragma unroll
    for (int k0 = 0; k0 < K; k0 += 32) {
        v16bf afrag;
        #pragma unroll
        for (int i = 0; i < 8; ++i) {
            const float2 fa = *(const float2*)(arow + k0 + ((i >> 2) << 4) + hi8 + ((i & 3) << 1));
            afrag[2 * i] = (__bf16)fa.x; afrag[2 * i + 1] = (__bf16)fa.y;
        }
        #pragma unroll
        for (int nt = 0; nt < NT; ++nt) {
            const float* wrow = W + (size_t)(tn + nt * 16 + r16) * K;
            v16bf bfrag;
            #pragma unroll
            for (int i = 0; i < 8; ++i) {
                const float2 fb = *(const float2*)(wrow + k0 + ((i >> 2) << 4) + hi8 + ((i & 3) << 1));
                bfrag[2 * i] = (__bf16)fb.x; bfrag[2 * i + 1] = (__bf16)fb.y;
            }
            acc[nt] = __builtin_amdgcn_wmma_f32_16x16x32_bf16(false, afrag, false, bfrag,
                                                              (short)0, acc[nt], false, false);
        }
    }
    #pragma unroll
    for (int nt = 0; nt < NT; ++nt) {
        #pragma unroll
        for (int r = 0; r < 8; ++r) {
            const int m = tm + r + hi8;
            const size_t idx = (size_t)m * N + tn + nt * 16 + r16;
            float vv = acc[nt][r];
            if (act) vv = gelu_f(vv);
            if (res) vv += res[idx];
            Y[idx] = vv;
        }
    }
}

// ---------------- transposes / residual adds ----------------
__global__ void transpose_pd(const float* __restrict__ in, float* __restrict__ out) {
    int idx = blockIdx.x * 256 + threadIdx.x;        // over BC*D*P, out layout (BC,D,P)
    int p = idx % Pp, d = (idx / Pp) % Dd, bc = idx / (Pp * Dd);
    out[idx] = in[((size_t)bc * Pp + p) * Dd + d];
}
__global__ void add_t_pd(float* __restrict__ xm, const float* __restrict__ t) {
    int idx = blockIdx.x * 256 + threadIdx.x;        // xm layout (BC,P,D)
    int d = idx % Dd, p = (idx / Dd) % Pp, bc = idx / (Dd * Pp);
    xm[idx] += t[((size_t)bc * Dd + d) * Pp + p];
}
__global__ void transpose_to_pdc(const float* __restrict__ in, float* __restrict__ out) {
    int idx = blockIdx.x * 256 + threadIdx.x;        // out layout (B,P,D,C)
    int c = idx % Cc, d = (idx / Cc) % Dd, p = (idx / (Cc * Dd)) % Pp, b = idx / (Cc * Dd * Pp);
    out[idx] = in[(((size_t)b * Cc + c) * Pp + p) * Dd + d];
}
__global__ void add_from_pdc(float* __restrict__ xm, const float* __restrict__ t) {
    int idx = blockIdx.x * 256 + threadIdx.x;        // xm layout (B,C,P,D)
    int d = idx % Dd, p = (idx / Dd) % Pp, c = (idx / (Dd * Pp)) % Cc, b = idx / (Dd * Pp * Cc);
    xm[idx] += t[(((size_t)b * Pp + p) * Dd + d) * Cc + c];
}

// ---------------- decoder ----------------
__global__ __launch_bounds__(256) void dec_aw_kernel(
    const float* __restrict__ y_mark, const float* __restrict__ scal,
    const float* __restrict__ dinv, const float* __restrict__ da_w1,
    const float* __restrict__ da_b1, const float* __restrict__ da_w2,
    const float* __restrict__ da_b2, float* __restrict__ aw_raw, int* __restrict__ ci) {
    int wave = (blockIdx.x * 256 + threadIdx.x) >> 5, lane = threadIdx.x & 31;
    if (wave >= Bb * PREDc * Kctx) return;
    int k = wave % Kctx, bj = wave / Kctx;
    float t = y_mark[bj], mint = scal[0], pw = scal[1];
    int qpi = (int)floorf((t - mint) / pw);
    qpi = min(max(qpi, 0), Pp - 1);
    int cik = min(max(qpi + k - KCc, 0), Pp - 1);
    float rel = t - (mint + pw * ((float)cik + 0.5f));
    float a = da_b1[lane];
    for (int e = 0; e < 32; ++e) {
        float fr = rel * dinv[e];
        a += da_w1[lane * Dhc + e] * sinf(fr) + da_w1[lane * Dhc + 32 + e] * cosf(fr);
    }
    float ps = wred_sum(gelu_f(a) * da_w2[lane]);
    if (lane == 0) { aw_raw[wave] = ps + da_b2[0]; ci[wave] = cik; }
}

__global__ void dec_softmax_kernel(float* __restrict__ aw) {
    int bj = blockIdx.x * 256 + threadIdx.x;
    if (bj >= Bb * PREDc) return;
    float m = -1e30f, e[Kctx], s = 0.f;
    for (int k = 0; k < Kctx; ++k) m = fmaxf(m, aw[bj * Kctx + k]);
    for (int k = 0; k < Kctx; ++k) { e[k] = expf(aw[bj * Kctx + k] - m); s += e[k]; }
    for (int k = 0; k < Kctx; ++k) aw[bj * Kctx + k] = e[k] / s;
}

__global__ void qte_kernel(const float* __restrict__ y_mark, const float* __restrict__ dinv,
                           float* __restrict__ qte) {
    int idx = blockIdx.x * 256 + threadIdx.x;        // over B*PRED*64
    int e = idx & 63, bj = idx >> 6;
    float t = y_mark[bj];
    qte[idx] = (e < 32) ? sinf(t * dinv[e]) : cosf(t * dinv[e - 32]);
}

__global__ void ctx_kernel(const float* __restrict__ xm_fin, const float* __restrict__ aw,
                           const int* __restrict__ ci, float* __restrict__ ctx) {
    int idx = blockIdx.x * 256 + threadIdx.x;        // ctx layout (B,C,PRED,D)
    int d = idx & 127, j = (idx >> 7) & (PREDc - 1), c = (idx >> 14) & (Cc - 1), b = idx >> 19;
    int bj = b * PREDc + j;
    float acc = 0.f;
    #pragma unroll
    for (int k = 0; k < Kctx; ++k)
        acc += aw[bj * Kctx + k] * xm_fin[(((size_t)b * Cc + c) * Pp + ci[bj * Kctx + k]) * Dd + d];
    ctx[idx] = acc;
}

// fused do_w1(gelu)+do_w2: one wave per (b,c,j) row
__global__ __launch_bounds__(256) void dec_out_kernel(
    const float* __restrict__ ctx, const float* __restrict__ qte,
    const float* __restrict__ do_w1, const float* __restrict__ do_b1,
    const float* __restrict__ do_w2, const float* __restrict__ do_b2,
    float* __restrict__ out) {
    int wave = (blockIdx.x * 256 + threadIdx.x) >> 5, lane = threadIdx.x & 31;
    if (wave >= Bb * Cc * PREDc) return;
    int b = wave / (Cc * PREDc), c = (wave / PREDc) % Cc, j = wave % PREDc;
    int bj = b * PREDc + j;
    const float* cr = ctx + (size_t)wave * Dd;
    const float* qr = qte + (size_t)bj * Dhc;
    float acc = 0.f;
    #pragma unroll 1
    for (int rr = 0; rr < 8; ++rr) {
        int h = lane + 32 * rr;
        const float* w1r = do_w1 + (size_t)h * (Dd + Dhc);
        float a = do_b1[h];
        for (int d = 0; d < Dd; ++d) a += w1r[d] * cr[d];
        for (int e = 0; e < Dhc; ++e) a += w1r[Dd + e] * qr[e];
        acc += gelu_f(a) * do_w2[h];
    }
    acc = wred_sum(acc);
    if (lane == 0) out[(size_t)(b * PREDc + j) * Cc + c] = acc + do_b2[0];
}

// =======================================================================
extern "C" void kernel_launch(void* const* d_in, const int* in_sizes, int n_in,
                              void* d_out, int out_size, void* d_ws, size_t ws_size,
                              hipStream_t stream) {
    // ---- inputs (setup_inputs dict order, nested dicts in insertion order) ----
    const float* x       = (const float*)d_in[0];
    const float* x_mark  = (const float*)d_in[1];
    const unsigned char* x_mask = (const unsigned char*)d_in[2];   // jnp bool -> 1 byte
    const float* y_mark  = (const float*)d_in[4];
    const float* inv_freq = (const float*)d_in[6];
    const float* dinv    = (const float*)d_in[7];
    const float* val_w   = (const float*)d_in[8];
    const float* val_b   = (const float*)d_in[9];
    const float* tm_w1   = (const float*)d_in[10];
    const float* tm_b1   = (const float*)d_in[11];
    const float* tm_w2   = (const float*)d_in[12];
    const float* tm_b2   = (const float*)d_in[13];
    const float* fus_w   = (const float*)d_in[14];
    const float* fus_b   = (const float*)d_in[15];
    const float* fus_g   = (const float*)d_in[16];
    const float* fus_be  = (const float*)d_in[17];
    const float* patch_q = (const float*)d_in[18];
    const float* wq      = (const float*)d_in[19];
    const float* bq      = (const float*)d_in[20];
    const float* wk      = (const float*)d_in[21];
    const float* bk      = (const float*)d_in[22];
    const float* wv      = (const float*)d_in[23];
    const float* bv      = (const float*)d_in[24];
    const float* wo      = (const float*)d_in[25];
    const float* bo      = (const float*)d_in[26];
    const float* empty   = (const float*)d_in[27];
    const float* fin_g   = (const float*)d_in[28];
    const float* fin_b   = (const float*)d_in[29];
    const float* da_w1   = (const float*)d_in[30];
    const float* da_b1   = (const float*)d_in[31];
    const float* da_w2   = (const float*)d_in[32];
    const float* da_b2   = (const float*)d_in[33];
    const float* do_w1   = (const float*)d_in[34];
    const float* do_b1   = (const float*)d_in[35];
    const float* do_w2   = (const float*)d_in[36];
    const float* do_b2   = (const float*)d_in[37];
    const float* ln1_g = (const float*)d_in[38]; const float* ln1_b = (const float*)d_in[39];
    const float* tt_w1 = (const float*)d_in[40]; const float* tt_b1 = (const float*)d_in[41];
    const float* tt_w2 = (const float*)d_in[42]; const float* tt_b2 = (const float*)d_in[43];
    const float* ln2_g = (const float*)d_in[44]; const float* ln2_b = (const float*)d_in[45];
    const float* tc_w1 = (const float*)d_in[46]; const float* tc_b1 = (const float*)d_in[47];
    const float* tc_w2 = (const float*)d_in[48]; const float* tc_b2 = (const float*)d_in[49];
    const float* ln3_g = (const float*)d_in[50]; const float* ln3_b = (const float*)d_in[51];
    const float* vt_w1 = (const float*)d_in[52]; const float* vt_b1 = (const float*)d_in[53];
    const float* vt_w2 = (const float*)d_in[54]; const float* vt_b2 = (const float*)d_in[55];
    const float* ln4_g = (const float*)d_in[56]; const float* ln4_b = (const float*)d_in[57];
    const float* vc_w1 = (const float*)d_in[58]; const float* vc_b1 = (const float*)d_in[59];
    const float* vc_w2 = (const float*)d_in[60]; const float* vc_b2 = (const float*)d_in[61];
    float* out = (float*)d_out;

    // ---- workspace bump allocator (~90 MB total) ----
    size_t off = 0;
    auto alloc = [&](size_t bytes) -> void* {
        void* p = (char*)d_ws + off;
        off += (bytes + 255) & ~(size_t)255;
        return p;
    };
    float* scal   = (float*)alloc(16 * 4);
    int*   startp = (int*)  alloc((size_t)Bb * (Pp + 1) * 4);
    float* u_vec  = (float*)alloc(Dd * 4);
    float* v_vec  = (float*)alloc(Dd * 4);
    float* c0buf  = (float*)alloc((size_t)Pp * Hh * 4);
    float* Aq     = (float*)alloc((size_t)Pp * Hh * Dd * 4);
    float* te_lin = (float*)alloc((size_t)Bb * Ll * Dd * 4);
    float* awbuf  = (float*)alloc((size_t)Bb * PREDc * Kctx * 4);
    int*   cibuf  = (int*)  alloc((size_t)Bb * PREDc * Kctx * 4);
    float* qtebuf = (float*)alloc((size_t)Bb * PREDc * Dhc * 4);
    float* xm     = (float*)alloc((size_t)Bb * Cc * Pp * Dd * 4);   // 16.8 MB
    float* bufA   = (float*)alloc((size_t)Bb * Cc * Pp * Dd * 4);   // 16.8 MB
    float* bufB   = (float*)alloc((size_t)Bb * Cc * Pp * Dd * 4);   // 16.8 MB
    float* bufC   = (float*)alloc((size_t)Bb * Cc * Dd * HPc * 4);  // 33.6 MB
    (void)n_in; (void)in_sizes; (void)out_size; (void)ws_size;

    const int elemPD = Bb * Cc * Pp * Dd;                 // 4,194,304
    const int rowsPD = Bb * Cc * Pp;                      // 32768

    // ---- stage 1: scalars, bounds, embedding pieces ----
    scalars_kernel<<<1, 256, 0, stream>>>(x_mark, y_mark, scal);
    bounds_kernel<<<(Bb * (Pp + 1) + 255) / 256, 256, 0, stream>>>(x_mark, scal, startp);
    uv_kernel<<<1, 128, 0, stream>>>(fus_w, fus_b, val_w, val_b, u_vec, v_vec);
    te_kernel<<<Bb * Ll, 128, 0, stream>>>(x_mark, inv_freq, tm_w1, tm_b1, tm_w2, tm_b2, fus_w, te_lin);
    qk_kernel<<<Pp * Hh, 128, 0, stream>>>(patch_q, wq, bq, wk, bk, Aq, c0buf);

    // ---- stage 2: segment-softmax attention pooling -> xm (B,C,P,D) ----
    pool_kernel<<<(Bb * Cc * Pp) / 8, 256, 0, stream>>>(
        x, x_mask, te_lin, u_vec, v_vec, fus_g, fus_be, Aq, c0buf,
        wv, bv, wo, bo, empty, startp, xm);

    // ---- stage 3: TAC-Mixer layers (WMMA bf16 GEMMs, 16x(16*NT) strips) ----
    #define GEMM(KK, NTT, Ain, Wp, Bp, Rp, Yp, MM, NN, ACT)                               \
        gemm_bf16_wmma<KK, NTT><<<((MM) / 16) * ((NN) / (16 * NTT)) / 8, 256, 0, stream>>>(\
            Ain, Wp, Bp, Rp, Yp, MM, NN, ACT)
    for (int i = 0; i < NLc; ++i) {
        // token-time mixing: (B,C,D,P), P->HP->P
        ln128_kernel<<<rowsPD / 8, 256, 0, stream>>>(xm, bufA, ln1_g + i * Dd, ln1_b + i * Dd, rowsPD);
        transpose_pd<<<elemPD / 256, 256, 0, stream>>>(bufA, bufB);
        GEMM(Pp,  4, bufB, tt_w1 + (size_t)i * HPc * Pp, tt_b1 + i * HPc, nullptr, bufC, Bb * Cc * Dd, HPc, 1);
        GEMM(HPc, 4, bufC, tt_w2 + (size_t)i * Pp * HPc, tt_b2 + i * Pp,  nullptr, bufB, Bb * Cc * Dd, Pp, 0);
        add_t_pd<<<elemPD / 256, 256, 0, stream>>>(xm, bufB);
        // channel mixing: D->DFF->D
        ln128_kernel<<<rowsPD / 8, 256, 0, stream>>>(xm, bufA, ln2_g + i * Dd, ln2_b + i * Dd, rowsPD);
        GEMM(Dd,   4, bufA, tc_w1 + (size_t)i * DFFc * Dd, tc_b1 + i * DFFc, nullptr, bufC, rowsPD, DFFc, 1);
        GEMM(DFFc, 4, bufC, tc_w2 + (size_t)i * Dd * DFFc, tc_b2 + i * Dd,   xm,      xm,   rowsPD, Dd, 0);
        // variable-token mixing: (B,P,D,C), C->HC->C
        ln128_kernel<<<rowsPD / 8, 256, 0, stream>>>(xm, bufA, ln3_g + i * Dd, ln3_b + i * Dd, rowsPD);
        transpose_to_pdc<<<elemPD / 256, 256, 0, stream>>>(bufA, bufB);
        GEMM(Cc,  4, bufB, vt_w1 + (size_t)i * HCc * Cc, vt_b1 + i * HCc, nullptr, bufC, Bb * Pp * Dd, HCc, 1);
        GEMM(HCc, 2, bufC, vt_w2 + (size_t)i * Cc * HCc, vt_b2 + i * Cc,  nullptr, bufB, Bb * Pp * Dd, Cc, 0);
        add_from_pdc<<<elemPD / 256, 256, 0, stream>>>(xm, bufB);
        // channel mixing 2
        ln128_kernel<<<rowsPD / 8, 256, 0, stream>>>(xm, bufA, ln4_g + i * Dd, ln4_b + i * Dd, rowsPD);
        GEMM(Dd,   4, bufA, vc_w1 + (size_t)i * DFFc * Dd, vc_b1 + i * DFFc, nullptr, bufC, rowsPD, DFFc, 1);
        GEMM(DFFc, 4, bufC, vc_w2 + (size_t)i * Dd * DFFc, vc_b2 + i * Dd,   xm,      xm,   rowsPD, Dd, 0);
    }
    // final LN -> bufA (= xm_fin)
    ln128_kernel<<<rowsPD / 8, 256, 0, stream>>>(xm, bufA, fin_g, fin_b, rowsPD);

    // ---- stage 4: decoder ----
    dec_aw_kernel<<<(Bb * PREDc * Kctx) / 8, 256, 0, stream>>>(
        y_mark, scal, dinv, da_w1, da_b1, da_w2, da_b2, awbuf, cibuf);
    dec_softmax_kernel<<<(Bb * PREDc + 255) / 256, 256, 0, stream>>>(awbuf);
    qte_kernel<<<(Bb * PREDc * Dhc) / 256, 256, 0, stream>>>(y_mark, dinv, qtebuf);
    ctx_kernel<<<(Bb * Cc * PREDc * Dd) / 256, 256, 0, stream>>>(bufA, awbuf, cibuf, bufC);
    dec_out_kernel<<<(Bb * Cc * PREDc) / 8, 256, 0, stream>>>(
        bufC, qtebuf, do_w1, do_b1, do_w2, do_b2, out);
    #undef GEMM
}
